// AAGNetSegmentor_19413252178389
// MI455X (gfx1250) — matrix operand order repaired and checked by reading, not compiled, not run on hardware
//
#include <hip/hip_runtime.h>
#include <hip/hip_bf16.h>
#include <math.h>

typedef __attribute__((ext_vector_type(16))) _Float16 v16h;
typedef __attribute__((ext_vector_type(8)))  float    v8f;

#define NTOK 1024
#define EDIM 128
#define NHEAD 8
#define HDIM 16
#define NLAYER 3
#define MLPH 256
#define HEADH 64
#define NCLS 25

// Pack two float4's (8 consecutive f32) into halves [base, base+8) of a v16h.
#define CVT8(frag, base, p, q)                                    \
    do {                                                          \
        (frag)[(base) + 0] = (_Float16)(p).x;                     \
        (frag)[(base) + 1] = (_Float16)(p).y;                     \
        (frag)[(base) + 2] = (_Float16)(p).z;                     \
        (frag)[(base) + 3] = (_Float16)(p).w;                     \
        (frag)[(base) + 4] = (_Float16)(q).x;                     \
        (frag)[(base) + 5] = (_Float16)(q).y;                     \
        (frag)[(base) + 6] = (_Float16)(q).z;                     \
        (frag)[(base) + 7] = (_Float16)(q).w;                     \
    } while (0)

// ---- CDNA5 fragment loaders (contiguous K order, same for A and B) ----
// A 16x32: lane lo holds row M=lo; halves h -> K=16*hi+h
__device__ __forceinline__ v16h load_a32(const float* __restrict__ Arow, int kk, int hi)
{
    v16h af;
    const float4* p0 = (const float4*)(Arow + kk + 16 * hi);
    float4 a0 = p0[0], a1 = p0[1], a2 = p0[2], a3 = p0[3];
    CVT8(af, 0, a0, a1);
    CVT8(af, 8, a2, a3);
    return af;
}
// A with K==16: hi half-wave (K=16..31) is zero
__device__ __forceinline__ v16h load_a16(const float* __restrict__ Arow, int hi)
{
    v16h af = {};
    if (!hi) {
        const float4* p = (const float4*)Arow;
        float4 a0 = p[0], a1 = p[1], a2 = p[2], a3 = p[3];
        CVT8(af, 0, a0, a1);
        CVT8(af, 8, a2, a3);
    }
    return af;
}
// B 32x16 from W[N][K] row-major: lane lo holds col N=lo; halves h -> K=16*hi+h
__device__ __forceinline__ v16h load_bT(const float* __restrict__ Brow, int kk, int hi)
{
    v16h bf;
    const float4* p = (const float4*)(Brow + kk + 16 * hi);
    float4 b0 = p[0], b1 = p[1], b2 = p[2], b3 = p[3];
    CVT8(bf, 0, b0, b1);
    CVT8(bf, 8, b2, b3);
    return bf;
}
// B with K==16: hi half-wave (K=16..31) is zero; row load always valid
__device__ __forceinline__ v16h load_bT16(const float* __restrict__ Brow, int hi)
{
    v16h bf;
    const float4* p = (const float4*)Brow;
    float4 b0 = p[0], b1 = p[1], b2 = p[2], b3 = p[3];
    CVT8(bf, 0, b0, b1);
    CVT8(bf, 8, b2, b3);
    if (hi) {
        #pragma unroll
        for (int h = 0; h < 16; ++h) bf[h] = (_Float16)0.0f;
    }
    return bf;
}
// B 32x16 from B[K][N] row-major (strided column gather)
__device__ __forceinline__ v16h load_bN(const float* __restrict__ B, int ldb,
                                        int kk, int hi, int wn)
{
    v16h bf;
    const float* bc = B + (size_t)(kk + 16 * hi) * ldb + wn;
    #pragma unroll
    for (int h = 0; h < 16; ++h) bf[h] = (_Float16)bc[(size_t)h * ldb];
    return bf;
}

// ---------------------------------------------------------------------------
// WMMA GEMM, register-blocked (MT x NT) 16x16 tiles per wave:
//   C[M,N] = act( scale * A[M,K] x B + bias ) (+ resid)
//  BMODE 0: B given as W[N][K] row-major (C = A @ W^T)
//  BMODE 1: B given as B[K][N] row-major (C = A @ B)
//  ACT   0: none   1: relu   2: gelu(exact)   3: += resid
//  KSPEC 0: requires M%(16*MT)==0, N%(16*NT)==0, K%32==0
//  KSPEC 16: K == 16 exactly (QK^T)
// MT*NT independent back-to-back v_wmma_f32_16x16x32_f16 per K-step.
// ---------------------------------------------------------------------------
template<int BMODE, int ACT, int KSPEC, int MT, int NT>
__global__ __launch_bounds__(32)
void wmma_gemm(const float* __restrict__ A, const float* __restrict__ B,
               const float* __restrict__ bias, const float* __restrict__ resid,
               float* __restrict__ C,
               int M, int N, int K, int lda, int ldb, int ldc, float scale)
{
    const int m0   = blockIdx.y * (16 * MT);
    const int n0   = blockIdx.x * (16 * NT);
    const int lane = threadIdx.x & 31;
    const int lo   = lane & 15;
    const int hi   = lane >> 4;

    v8f acc[MT][NT];
    #pragma unroll
    for (int mt = 0; mt < MT; ++mt)
        #pragma unroll
        for (int nt = 0; nt < NT; ++nt) acc[mt][nt] = (v8f){};

    const float* Arow[MT];
    #pragma unroll
    for (int mt = 0; mt < MT; ++mt)
        Arow[mt] = A + (size_t)(m0 + mt * 16 + lo) * lda;
    const float* Brow[NT];   // BMODE 0 only
    #pragma unroll
    for (int nt = 0; nt < NT; ++nt)
        Brow[nt] = B + (size_t)(n0 + nt * 16 + lo) * ldb;

    if (KSPEC == 16) {
        v16h af[MT], bf[NT];
        #pragma unroll
        for (int mt = 0; mt < MT; ++mt) af[mt] = load_a16(Arow[mt], hi);
        #pragma unroll
        for (int nt = 0; nt < NT; ++nt) bf[nt] = load_bT16(Brow[nt], hi);
        #pragma unroll
        for (int mt = 0; mt < MT; ++mt)
            #pragma unroll
            for (int nt = 0; nt < NT; ++nt)
                acc[mt][nt] = __builtin_amdgcn_wmma_f32_16x16x32_f16(
                    false, af[mt], false, bf[nt], (short)0, acc[mt][nt], false, false);
    } else {
        for (int kk = 0; kk < K; kk += 32) {
            v16h af[MT], bf[NT];
            #pragma unroll
            for (int mt = 0; mt < MT; ++mt) af[mt] = load_a32(Arow[mt], kk, hi);
            #pragma unroll
            for (int nt = 0; nt < NT; ++nt) {
                if (BMODE == 0) bf[nt] = load_bT(Brow[nt], kk, hi);
                else            bf[nt] = load_bN(B, ldb, kk, hi, n0 + nt * 16 + lo);
            }
            #pragma unroll
            for (int mt = 0; mt < MT; ++mt)
                #pragma unroll
                for (int nt = 0; nt < NT; ++nt)
                    acc[mt][nt] = __builtin_amdgcn_wmma_f32_16x16x32_f16(
                        false, af[mt], false, bf[nt], (short)0, acc[mt][nt], false, false);
        }
    }

    // ---- epilogue: lane lo holds column n; vgpr r holds row m = 8*hi + r ----
    #pragma unroll
    for (int nt = 0; nt < NT; ++nt) {
        const int col = n0 + nt * 16 + lo;
        const float bv = bias ? bias[col] : 0.0f;
        #pragma unroll
        for (int mt = 0; mt < MT; ++mt) {
            #pragma unroll
            for (int r = 0; r < 8; ++r) {
                const int row = m0 + mt * 16 + hi * 8 + r;
                float v = acc[mt][nt][r] * scale + bv;
                if (ACT == 1) v = fmaxf(v, 0.0f);
                else if (ACT == 2) v = 0.5f * v * (1.0f + erff(v * 0.70710678118654752f));
                else if (ACT == 3) v += resid[(size_t)row * ldc + col];
                C[(size_t)row * ldc + col] = v;
            }
        }
    }
}

// ---------------------------------------------------------------------------
// Input embedding: feat[n, 0:64] = x[:, :10] @ node_w^T + node_b
//                  feat[n,64:128]= x[:, 10:] @ grid_w^T + grid_b
// ---------------------------------------------------------------------------
__global__ void embed_kernel(const float* __restrict__ x,
                             const float* __restrict__ nw, const float* __restrict__ nb,
                             const float* __restrict__ gw, const float* __restrict__ gb,
                             float* __restrict__ feat)
{
    int row = blockIdx.x;
    int c   = threadIdx.x;            // 0..127
    const float* xr = x + (size_t)row * 17;
    float v;
    if (c < 64) {
        v = nb[c];
        #pragma unroll
        for (int k = 0; k < 10; ++k) v = fmaf(xr[k], nw[c * 10 + k], v);
    } else {
        int cc = c - 64;
        v = gb[cc];
        #pragma unroll
        for (int k = 0; k < 7; ++k) v = fmaf(xr[10 + k], gw[cc * 7 + k], v);
    }
    feat[(size_t)row * EDIM + c] = v;
}

// ---------------------------------------------------------------------------
// Row softmax over ncols columns; one block (256 threads) per row, in place.
// ---------------------------------------------------------------------------
__global__ void softmax_rows(float* __restrict__ S, int ncols)
{
    __shared__ float red[256];
    const int t = threadIdx.x;
    float* p = S + (size_t)blockIdx.x * ncols;

    float mx = -INFINITY;
    for (int j = t; j < ncols; j += 256) mx = fmaxf(mx, p[j]);
    red[t] = mx; __syncthreads();
    for (int s = 128; s > 0; s >>= 1) { if (t < s) red[t] = fmaxf(red[t], red[t + s]); __syncthreads(); }
    mx = red[0]; __syncthreads();

    float sum = 0.0f;
    for (int j = t; j < ncols; j += 256) { float e = __expf(p[j] - mx); p[j] = e; sum += e; }
    red[t] = sum; __syncthreads();
    for (int s = 128; s > 0; s >>= 1) { if (t < s) red[t] += red[t + s]; __syncthreads(); }
    const float inv = 1.0f / red[0];
    __syncthreads();
    for (int j = t; j < ncols; j += 256) p[j] *= inv;
}

// ---------------------------------------------------------------------------
// feat = LayerNorm(residual + delta) * w + b    (one block of 128 per row)
// ---------------------------------------------------------------------------
__global__ void add_layernorm(const float* __restrict__ residual,
                              const float* __restrict__ delta,
                              const float* __restrict__ w, const float* __restrict__ b,
                              float* __restrict__ out)
{
    __shared__ float red[128];
    const int row = blockIdx.x, t = threadIdx.x;
    const size_t idx = (size_t)row * EDIM + t;
    float x = residual[idx] + delta[idx];
    red[t] = x; __syncthreads();
    for (int s = 64; s > 0; s >>= 1) { if (t < s) red[t] += red[t + s]; __syncthreads(); }
    const float mu = red[0] * (1.0f / EDIM); __syncthreads();
    const float d = x - mu;
    red[t] = d * d; __syncthreads();
    for (int s = 64; s > 0; s >>= 1) { if (t < s) red[t] += red[t + s]; __syncthreads(); }
    const float var = red[0] * (1.0f / EDIM);
    out[idx] = d * rsqrtf(var + 1e-5f) * w[t] + b[t];
}

// ---------------------------------------------------------------------------
// Small output heads: out[row, c] = Ain[row,:K] . W[c,:K] + b[c]
// ---------------------------------------------------------------------------
__global__ void small_head(const float* __restrict__ Ain, const float* __restrict__ W,
                           const float* __restrict__ b, float* __restrict__ out,
                           int ncols, int K, int ldout)
{
    int row = blockIdx.x, c = threadIdx.x;
    if (c >= ncols) return;
    float v = b[c];
    const float* ar = Ain + (size_t)row * K;
    const float* wr = W + (size_t)c * K;
    for (int k = 0; k < K; ++k) v = fmaf(ar[k], wr[k], v);
    out[(size_t)row * ldout + c] = v;
}

// ---------------------------------------------------------------------------
// Instance matrix:  out[i,j] = (i==j) ? 0
//     : sum_h relu(a[min,h] + b[max,h] + bias[h]) * w2[h] + b2
// (equals triu(M,1) + triu(M,1)^T of M[i,j] = relu(a_i+b_j+bias).w2 + b2)
// 16x16 output tile per block; LDS rows padded to 65 floats (no bank conflicts)
// ---------------------------------------------------------------------------
__global__ void pair_inst(const float* __restrict__ a, const float* __restrict__ bvec,
                          const float* __restrict__ bias, const float* __restrict__ w2,
                          const float* __restrict__ b2p, float* __restrict__ out)
{
    __shared__ float aI[16][65], aJ[16][65], bI[16][65], bJ[16][65];
    __shared__ float sb[64], sw[64];
    const int i0 = blockIdx.y * 16, j0 = blockIdx.x * 16;
    const int t = threadIdx.y * 16 + threadIdx.x;   // 0..255

    for (int u = t; u < 16 * 64; u += 256) {
        int r = u >> 6, c = u & 63;
        aI[r][c] = a[(size_t)(i0 + r) * HEADH + c];
        aJ[r][c] = a[(size_t)(j0 + r) * HEADH + c];
        bI[r][c] = bvec[(size_t)(i0 + r) * HEADH + c];
        bJ[r][c] = bvec[(size_t)(j0 + r) * HEADH + c];
    }
    if (t < 64) { sb[t] = bias[t]; sw[t] = w2[t]; }
    __syncthreads();

    const int ti = threadIdx.y, tj = threadIdx.x;
    const int i = i0 + ti, j = j0 + tj;
    float acc = 0.0f;
    if (i < j) {
        #pragma unroll 8
        for (int h = 0; h < 64; ++h) {
            float v = fmaxf(aI[ti][h] + bJ[tj][h] + sb[h], 0.0f);
            acc = fmaf(v, sw[h], acc);
        }
        acc += b2p[0];
    } else if (i > j) {
        #pragma unroll 8
        for (int h = 0; h < 64; ++h) {
            float v = fmaxf(aJ[tj][h] + bI[ti][h] + sb[h], 0.0f);
            acc = fmaf(v, sw[h], acc);
        }
        acc += b2p[0];
    }
    out[(size_t)i * NTOK + j] = acc;
}

// ---------------------------------------------------------------------------
extern "C" void kernel_launch(void* const* d_in, const int* in_sizes, int n_in,
                              void* d_out, int out_size, void* d_ws, size_t ws_size,
                              hipStream_t stream)
{
    const float* x         = (const float*)d_in[0];
    const float* node_w    = (const float*)d_in[1];
    const float* node_b    = (const float*)d_in[2];
    const float* grid_w    = (const float*)d_in[3];
    const float* grid_b    = (const float*)d_in[4];
    const float* in_proj_w = (const float*)d_in[5];
    const float* in_proj_b = (const float*)d_in[6];
    const float* out_w     = (const float*)d_in[7];
    const float* out_b     = (const float*)d_in[8];
    const float* ln_w      = (const float*)d_in[9];
    const float* ln_b      = (const float*)d_in[10];
    const float* mlp_w1    = (const float*)d_in[11];
    const float* mlp_b1    = (const float*)d_in[12];
    const float* mlp_w2    = (const float*)d_in[13];
    const float* mlp_b2    = (const float*)d_in[14];
    const float* seg_w1    = (const float*)d_in[15];
    const float* seg_b1    = (const float*)d_in[16];
    const float* seg_w2    = (const float*)d_in[17];
    const float* seg_b2    = (const float*)d_in[18];
    const float* inst_w1   = (const float*)d_in[19];
    const float* inst_b1   = (const float*)d_in[20];
    const float* inst_w2   = (const float*)d_in[21];
    const float* inst_b2   = (const float*)d_in[22];
    const float* bot_w1    = (const float*)d_in[23];
    const float* bot_b1    = (const float*)d_in[24];
    const float* bot_w2    = (const float*)d_in[25];
    const float* bot_b2    = (const float*)d_in[26];

    // Workspace layout (float element offsets)
    float* ws    = (float*)d_ws;
    float* feat  = ws;                        // 1024*128
    float* qkv   = feat  + NTOK * EDIM;       // 1024*384
    float* sc    = qkv   + NTOK * 3 * EDIM;   // 1024*1024 (per-head, reused)
    float* atto  = sc    + NTOK * NTOK;       // 1024*128
    float* prj   = atto  + NTOK * EDIM;       // 1024*128
    float* mid   = prj   + NTOK * EDIM;       // 1024*256
    float* hbuf  = mid   + NTOK * MLPH;       // 1024*64 (seg/bot hidden)
    float* abuf  = hbuf  + NTOK * HEADH;      // 1024*64
    float* bbuf  = abuf  + NTOK * HEADH;      // 1024*64

    float* out_seg  = (float*)d_out;                    // 1024*25
    float* out_inst = out_seg + NTOK * NCLS;            // 1024*1024
    float* out_bot  = out_inst + (size_t)NTOK * NTOK;   // 1024

    const dim3 wave(32);
    // grid for MTxNT register-blocked gemm
    auto tiles22 = [](int M, int N) { return dim3(N / 32, M / 32); };
    auto tiles21 = [](int M, int N) { return dim3(N / 16, M / 32); };
    const float inv_sqrt_hd = 0.25f;  // 1/sqrt(16)

    // ---- input embedding ----
    embed_kernel<<<NTOK, EDIM, 0, stream>>>(x, node_w, node_b, grid_w, grid_b, feat);

    // ---- transformer layers ----
    for (int l = 0; l < NLAYER; ++l) {
        const float* Wqkv = in_proj_w + (size_t)l * 3 * EDIM * EDIM;
        const float* Bqkv = in_proj_b + (size_t)l * 3 * EDIM;

        // qkv = feat @ Wqkv^T + b   (1024 x 384)
        wmma_gemm<0, 0, 0, 2, 2><<<tiles22(NTOK, 3 * EDIM), wave, 0, stream>>>(
            feat, Wqkv, Bqkv, nullptr, qkv,
            NTOK, 3 * EDIM, EDIM, EDIM, EDIM, 3 * EDIM, 1.0f);

        // per-head attention, score buffer reused sequentially
        for (int h = 0; h < NHEAD; ++h) {
            const float* qh = qkv + h * HDIM;                 // lda = 384
            const float* kh = qkv + EDIM + h * HDIM;
            const float* vh = qkv + 2 * EDIM + h * HDIM;
            // scores = q @ k^T / sqrt(hd)   (K = 16, zero padded to 32)
            wmma_gemm<0, 0, 16, 2, 2><<<tiles22(NTOK, NTOK), wave, 0, stream>>>(
                qh, kh, nullptr, nullptr, sc,
                NTOK, NTOK, HDIM, 3 * EDIM, 3 * EDIM, NTOK, inv_sqrt_hd);
            softmax_rows<<<NTOK, 256, 0, stream>>>(sc, NTOK);
            // o_head = attn @ v   (B is [K=1024][N=16] row-major, ldb=384)
            wmma_gemm<1, 0, 0, 2, 1><<<tiles21(NTOK, HDIM), wave, 0, stream>>>(
                sc, vh, nullptr, nullptr, atto + h * HDIM,
                NTOK, HDIM, NTOK, NTOK, 3 * EDIM, EDIM, 1.0f);
        }

        // out projection
        wmma_gemm<0, 0, 0, 2, 2><<<tiles22(NTOK, EDIM), wave, 0, stream>>>(
            atto, out_w + (size_t)l * EDIM * EDIM, out_b + (size_t)l * EDIM,
            nullptr, prj, NTOK, EDIM, EDIM, EDIM, EDIM, EDIM, 1.0f);

        // feat = LN(feat + prj)
        add_layernorm<<<NTOK, EDIM, 0, stream>>>(
            feat, prj, ln_w + (size_t)l * EDIM, ln_b + (size_t)l * EDIM, feat);

        // mid = gelu(feat @ mlp_w1^T + b1)
        wmma_gemm<0, 2, 0, 2, 2><<<tiles22(NTOK, MLPH), wave, 0, stream>>>(
            feat, mlp_w1 + (size_t)l * MLPH * EDIM, mlp_b1 + (size_t)l * MLPH,
            nullptr, mid, NTOK, MLPH, EDIM, EDIM, EDIM, MLPH, 1.0f);

        // feat = feat + mid @ mlp_w2^T + b2
        wmma_gemm<0, 3, 0, 2, 2><<<tiles22(NTOK, EDIM), wave, 0, stream>>>(
            mid, mlp_w2 + (size_t)l * EDIM * MLPH, mlp_b2 + (size_t)l * EDIM,
            feat, feat, NTOK, EDIM, MLPH, MLPH, MLPH, EDIM, 1.0f);
    }

    // ---- segmentation head ----
    wmma_gemm<0, 1, 0, 2, 2><<<tiles22(NTOK, HEADH), wave, 0, stream>>>(
        feat, seg_w1, seg_b1, nullptr, hbuf,
        NTOK, HEADH, EDIM, EDIM, EDIM, HEADH, 1.0f);
    small_head<<<NTOK, 32, 0, stream>>>(hbuf, seg_w2, seg_b2, out_seg, NCLS, HEADH, NCLS);

    // ---- instance head ----
    wmma_gemm<0, 0, 0, 2, 2><<<tiles22(NTOK, HEADH), wave, 0, stream>>>(
        feat, inst_w1, nullptr, nullptr, abuf,                 // Wa = inst_w1[:, :128]
        NTOK, HEADH, EDIM, EDIM, 2 * EDIM, HEADH, 1.0f);
    wmma_gemm<0, 0, 0, 2, 2><<<tiles22(NTOK, HEADH), wave, 0, stream>>>(
        feat, inst_w1 + EDIM, nullptr, nullptr, bbuf,          // Wb = inst_w1[:, 128:]
        NTOK, HEADH, EDIM, EDIM, 2 * EDIM, HEADH, 1.0f);
    pair_inst<<<dim3(NTOK / 16, NTOK / 16), dim3(16, 16), 0, stream>>>(
        abuf, bbuf, inst_b1, inst_w2, inst_b2, out_inst);

    // ---- bottom head ----
    wmma_gemm<0, 1, 0, 2, 2><<<tiles22(NTOK, HEADH), wave, 0, stream>>>(
        feat, bot_w1, bot_b1, nullptr, hbuf,
        NTOK, HEADH, EDIM, EDIM, EDIM, HEADH, 1.0f);
    small_head<<<NTOK, 32, 0, stream>>>(hbuf, bot_w2, bot_b2, out_bot, 1, HEADH, 1);
}
